// LightGCN_14001593385335
// MI455X (gfx1250) — compile-verified, matrix operand order checked by reading, not run on hardware
//
#include <hip/hip_runtime.h>
#include <hip/hip_bf16.h>

#define N_USERS 100000
#define N_ITEMS 200000
#define N_NODES (N_USERS + N_ITEMS)
#define BATCH   16384
#define DIM     64

#define NODE_ELEMS  ((size_t)N_NODES * DIM)   // 19.2M floats = 76.8 MB
#define BATCH_ELEMS ((size_t)BATCH * DIM)     // 1.05M floats = 4.2 MB

typedef __attribute__((ext_vector_type(2)))  float    v2f;
typedef __attribute__((ext_vector_type(8)))  float    v8f;
typedef __attribute__((ext_vector_type(16))) _Float16 v16h;

// ---------------------------------------------------------------------------
// 1) x0 = concat(user_emb, item_emb); x1 = 0
// ---------------------------------------------------------------------------
__global__ void k_init_x(const float* __restrict__ uemb, const float* __restrict__ iemb,
                         float* __restrict__ x0, float* __restrict__ x1) {
  size_t i = (size_t)blockIdx.x * blockDim.x + threadIdx.x;
  if (i >= NODE_ELEMS) return;
  const size_t uel = (size_t)N_USERS * DIM;
  x0[i] = (i < uel) ? uemb[i] : iemb[i - uel];
  x1[i] = 0.0f;
}

// ---------------------------------------------------------------------------
// 2) seed per-pair accumulators with layer-0 embedding rows
// ---------------------------------------------------------------------------
__global__ void k_init_acc(const int* __restrict__ users, const int* __restrict__ items,
                           const float* __restrict__ uemb, const float* __restrict__ iemb,
                           float* __restrict__ ua, float* __restrict__ ia) {
  int i = blockIdx.x * blockDim.x + threadIdx.x;
  if (i >= (int)BATCH_ELEMS) return;
  int b = i >> 6, d = i & 63;
  ua[i] = uemb[(size_t)users[b] * DIM + d];
  ia[i] = iemb[(size_t)items[b] * DIM + d];
}

// ---------------------------------------------------------------------------
// 3) scatter: one wave32 per edge; lanes stripe the 64-float row as float2.
//    Gathers and atomics both target the 76.8MB x buffers -> L2 resident.
// ---------------------------------------------------------------------------
__global__ void k_scatter(const int* __restrict__ esrc, const int* __restrict__ edst,
                          const float* __restrict__ eval, const float* __restrict__ xin,
                          float* __restrict__ xout, int nnz) {
  int wave = (int)(((size_t)blockIdx.x * blockDim.x + threadIdx.x) >> 5);
  int lane = threadIdx.x & 31;
  if (wave >= nnz) return;
  int   s = esrc[wave];
  int   d = edst[wave];
  float v = eval[wave];
  const v2f* pin = (const v2f*)(xin + (size_t)s * DIM);
  v2f t = pin[lane];                       // global_load_b64, coalesced 256B row
  float* po = xout + (size_t)d * DIM + 2 * lane;
  unsafeAtomicAdd(po,     v * t.x);        // global_atomic_add_f32 (no return)
  unsafeAtomicAdd(po + 1, v * t.y);
}

// ---------------------------------------------------------------------------
// 4) after each layer: fold this layer's rows into the batch accumulators
// ---------------------------------------------------------------------------
__global__ void k_gather_acc(const int* __restrict__ users, const int* __restrict__ items,
                             const float* __restrict__ x,
                             float* __restrict__ ua, float* __restrict__ ia) {
  int i = blockIdx.x * blockDim.x + threadIdx.x;
  if (i >= (int)BATCH_ELEMS) return;
  int b = i >> 6, d = i & 63;
  ua[i] += x[(size_t)users[b] * DIM + d];
  ia[i] += x[((size_t)N_USERS + (size_t)items[b]) * DIM + d];
}

__global__ void k_zero(float* __restrict__ x) {
  size_t i = (size_t)blockIdx.x * blockDim.x + threadIdx.x;
  if (i < NODE_ELEMS) x[i] = 0.0f;
}

// ---------------------------------------------------------------------------
// 5) batched dot via WMMA: per wave, a 16-pair tile. gamma = diag(U·I^T)/16.
//    f32 path: 16 x V_WMMA_F32_16X16X4_F32 chained on the f32 accumulator.
//    A 16x4 layout: lanes 0-15 hold (K=0,K=1), lanes 16-31 hold (K=2,K=3).
//    D tile spilled to LDS to extract the diagonal.
// ---------------------------------------------------------------------------
__global__ void k_dot_wmma(const float* __restrict__ ua, const float* __restrict__ ia,
                           float* __restrict__ out) {
  __shared__ float tile[8][16 * 16];
  int wave = threadIdx.x >> 5;                 // 0..7
  int lane = threadIdx.x & 31;
  int tileIdx = blockIdx.x * 8 + wave;         // 0..BATCH/16-1
  int row  = (tileIdx << 4) + (lane & 15);     // batch pair covered by this lane
  const float* up = ua + (size_t)row * DIM;
  const float* ip = ia + (size_t)row * DIM;

  v8f c = {};
#if __has_builtin(__builtin_amdgcn_wmma_f32_16x16x4_f32)
  int koff = (lane >> 4) << 1;                 // 0 or 2
#pragma unroll
  for (int ch = 0; ch < 16; ++ch) {
    int k = ch * 4 + koff;
    v2f a; a.x = up[k]; a.y = up[k + 1];       // A[m][k]   = U[row][k]
    v2f b; b.x = ip[k]; b.y = ip[k + 1];       // B[k][n]   = I[col][k] (I^T)
    c = __builtin_amdgcn_wmma_f32_16x16x4_f32(
        false, a, false, b, (short)0, c, false, false);
  }
#else
  // Fallback: codegen-confirmed f16 WMMA (accumulate still f32).
  int base_lo = (lane < 16) ? 0 : 8;
  int base_hi = (lane < 16) ? 16 : 24;
#pragma unroll
  for (int ch = 0; ch < 2; ++ch) {
    int k0 = ch * 32;
    v16h a, b;
#pragma unroll
    for (int h = 0; h < 8; ++h) {
      a[h]     = (_Float16)up[k0 + base_lo + h];
      a[h + 8] = (_Float16)up[k0 + base_hi + h];
      b[h]     = (_Float16)ip[k0 + base_lo + h];
      b[h + 8] = (_Float16)ip[k0 + base_hi + h];
    }
    c = __builtin_amdgcn_wmma_f32_16x16x32_f16(
        false, a, false, b, (short)0, c, false, false);
  }
#endif

  // D layout: VGPR r: lanes 0-15 -> (M=r, N=lane); lanes 16-31 -> (M=r+8, N=lane-16)
  float* t = tile[wave];
  int n = lane & 15;
  int mbase = (lane >> 4) << 3;
#pragma unroll
  for (int r = 0; r < 8; ++r) t[(mbase + r) * 16 + n] = c[r];
  __syncthreads();
  if (lane < 16) {
    // acc/4 on each side -> product scaled by 1/16
    out[(tileIdx << 4) + lane] = t[lane * 16 + lane] * (1.0f / 16.0f);
  }
}

// ---------------------------------------------------------------------------
// launcher
// ---------------------------------------------------------------------------
extern "C" void kernel_launch(void* const* d_in, const int* in_sizes, int n_in,
                              void* d_out, int out_size, void* d_ws, size_t ws_size,
                              hipStream_t stream) {
  (void)n_in; (void)out_size; (void)ws_size;
  const int*   users = (const int*)d_in[0];
  const int*   items = (const int*)d_in[1];
  const int*   esrc  = (const int*)d_in[2];
  const int*   edst  = (const int*)d_in[3];
  const float* eval  = (const float*)d_in[4];
  const float* uemb  = (const float*)d_in[5];
  const float* iemb  = (const float*)d_in[6];
  float* gamma = (float*)d_out;
  const int nnz = in_sizes[2];

  float* x0 = (float*)d_ws;              // 76.8 MB
  float* x1 = x0 + NODE_ELEMS;           // 76.8 MB
  float* ua = x1 + NODE_ELEMS;           //  4.2 MB
  float* ia = ua + BATCH_ELEMS;          //  4.2 MB

  const int TH = 256;
  const unsigned blNode  = (unsigned)((NODE_ELEMS  + TH - 1) / TH);
  const unsigned blBatch = (unsigned)((BATCH_ELEMS + TH - 1) / TH);

  k_init_x  <<<blNode,  TH, 0, stream>>>(uemb, iemb, x0, x1);
  k_init_acc<<<blBatch, TH, 0, stream>>>(users, items, uemb, iemb, ua, ia);

  float* xin = x0;
  float* xout = x1;
  for (int l = 0; l < 3; ++l) {
    unsigned blEdge = (unsigned)((nnz + 7) / 8);   // 8 waves (edges) per block
    k_scatter   <<<blEdge,  TH, 0, stream>>>(esrc, edst, eval, xin, xout, nnz);
    k_gather_acc<<<blBatch, TH, 0, stream>>>(users, items, xout, ua, ia);
    if (l < 2) k_zero<<<blNode, TH, 0, stream>>>(xin);  // becomes next xout
    float* tmp = xin; xin = xout; xout = tmp;
  }

  // 16384 pairs / 16 per wave / 8 waves per block = 128 blocks
  k_dot_wmma<<<BATCH / 128, TH, 0, stream>>>(ua, ia, gamma);
}
